// GATNet_738734375045
// MI455X (gfx1250) — matrix operand order repaired and checked by reading, not compiled, not run on hardware
//
#include <hip/hip_runtime.h>
#include <hip/hip_bf16.h>

// ---------------------------------------------------------------------------
// GATNet for MI455X (gfx1250): WMMA f16 GEMMs + atomic segment-softmax passes
// GEMM: one wave computes a 16x64 C strip (4 WMMA tiles) so each A fragment
// load feeds 4 v_wmma_f32_16x16x32_f16 ops; K templated for full unroll.
// ---------------------------------------------------------------------------

typedef _Float16 half_t;
typedef __attribute__((ext_vector_type(16))) _Float16 v16h;
typedef __attribute__((ext_vector_type(8)))  _Float16 v8h;
typedef __attribute__((ext_vector_type(8)))  float    v8f;

#define NN     50000
#define FIN    256
#define HEADS  4
#define HID    64
#define HC1    (HEADS * HID)   // 256
#define EORIG  800000
#define ETOT   (EORIG + NN)    // 850000 (self loops appended)
#define NG     512

// ------------------------- helpers -----------------------------------------

__device__ __forceinline__ unsigned fenc(float f) {
    unsigned u = __float_as_uint(f);
    return (u & 0x80000000u) ? ~u : (u | 0x80000000u);
}
__device__ __forceinline__ float fdec(unsigned u) {
    unsigned b = (u & 0x80000000u) ? (u & 0x7FFFFFFFu) : ~u;
    return __uint_as_float(b);
}
__device__ __forceinline__ float elu_f(float x) {
    return x > 0.0f ? x : (__expf(x) - 1.0f);
}

// ------------------------- conversion kernels -------------------------------

__global__ void cvt_f32_to_f16(const float* __restrict__ src,
                               half_t* __restrict__ dst, int n) {
    int i = blockIdx.x * blockDim.x + threadIdx.x;
    if (i < n) dst[i] = (half_t)src[i];
}

// Wt[n][k] = (f16) W[k][n]   (W is [K,N] row-major)
__global__ void transpose_to_f16(const float* __restrict__ W,
                                 half_t* __restrict__ Wt, int K, int N) {
    int i = blockIdx.x * blockDim.x + threadIdx.x;
    if (i >= K * N) return;
    int n = i / K, k = i - n * K;
    Wt[(size_t)n * K + k] = (half_t)W[(size_t)k * N + n];
}

// ------------------------- WMMA GEMM ----------------------------------------
// C[M,N] = A[M,K](f16,row-major) * Bt[N,K](f16, B transposed) ; f32 accum.
// One wave per 16x(16*NT) C strip: the A fragment is loaded once per K-step
// and reused by NT consecutive WMMA tiles along N.

template<int K, int NT>
__global__ __launch_bounds__(32)
void wmma_gemm_f16(const half_t* __restrict__ A, const half_t* __restrict__ Bt,
                   float* __restrict__ C, int N) {
    const int lane  = threadIdx.x & 31;
    const int mt    = blockIdx.x;
    const int nt0   = blockIdx.y * NT;           // first 16-col tile of strip
    const int l15   = lane & 15;
    const int hiHlf = lane >> 4;                 // 0 or 1 (lane group)
    // A frag: lane holds row m, K = ka..ka+15 (ka = 0|16)
    const half_t* ap = A  + (size_t)(mt * 16 + l15) * K + (size_t)(hiHlf * 16);
    // B frag: lane holds col n, K = kb..kb+15 (kb = 0|16)
    const half_t* bp = Bt + (size_t)(nt0 * 16 + l15) * K + (size_t)(hiHlf * 16);

    v8f acc[NT];
#pragma unroll
    for (int j = 0; j < NT; ++j) acc[j] = (v8f){};

#pragma unroll
    for (int k0 = 0; k0 < K; k0 += 32) {
        v8h alo = *(const v8h*)(ap + k0);
        v8h ahi = *(const v8h*)(ap + k0 + 8);
        v16h av = __builtin_shufflevector(alo, ahi,
                    0,1,2,3,4,5,6,7,8,9,10,11,12,13,14,15);
#pragma unroll
        for (int j = 0; j < NT; ++j) {
            const half_t* bpj = bp + (size_t)j * (16 * K);
            v8h blo = *(const v8h*)(bpj + k0);
            v8h bhi = *(const v8h*)(bpj + k0 + 8);
            v16h bv = __builtin_shufflevector(blo, bhi,
                        0,1,2,3,4,5,6,7,8,9,10,11,12,13,14,15);
            acc[j] = __builtin_amdgcn_wmma_f32_16x16x32_f16(
                false, av, false, bv, (short)0, acc[j], false, false);
        }
    }
    // C/D layout: VGPR r -> row mt*16 + r + 8*hiHlf ; col tile*16 + (lane&15)
    const int rbase = mt * 16 + hiHlf * 8;
#pragma unroll
    for (int j = 0; j < NT; ++j) {
        const int col = (nt0 + j) * 16 + l15;
#pragma unroll
        for (int r = 0; r < 8; ++r)
            C[(size_t)(rbase + r) * N + col] = acc[j][r];
    }
}

// ------------------------- GAT attention kernels ----------------------------

// asrc[n,h] = sum_c hlin[n, h*64+c] * att_src[h,c]  (and same for dst)
__global__ void node_att_dots(const float* __restrict__ hlin,
                              const float* __restrict__ att_src,
                              const float* __restrict__ att_dst,
                              float* __restrict__ asrc, float* __restrict__ adst,
                              int H) {
    int i = blockIdx.x * blockDim.x + threadIdx.x;   // i = n*H + h
    if (i >= NN * H) return;
    int n = i / H, h = i - n * H;
    const float* hp = hlin + (size_t)n * (H * 64) + h * 64;
    const float* as = att_src + h * 64;
    const float* ad = att_dst + h * 64;
    float ss = 0.0f, ds = 0.0f;
#pragma unroll 8
    for (int c = 0; c < 64; ++c) { ss += hp[c] * as[c]; ds += hp[c] * ad[c]; }
    asrc[i] = ss; adst[i] = ds;
}

__device__ __forceinline__ void edge_sd(const int* __restrict__ ei, int e,
                                        int& s, int& d) {
    if (e < EORIG) { s = ei[e]; d = ei[EORIG + e]; }
    else           { s = e - EORIG; d = s; }        // self loop
}

// Pass A: e = leaky_relu(asrc[s]+adst[d], 0.2); segment max over dst (encoded)
__global__ void edge_logits(const int* __restrict__ ei,
                            const float* __restrict__ asrc,
                            const float* __restrict__ adst,
                            float* __restrict__ ebuf,
                            unsigned* __restrict__ emax, int H) {
    int e = blockIdx.x * blockDim.x + threadIdx.x;
    if (e >= ETOT) return;
    int s, d; edge_sd(ei, e, s, d);
    for (int h = 0; h < H; ++h) {
        float v = asrc[(size_t)s * H + h] + adst[(size_t)d * H + h];
        v = v > 0.0f ? v : 0.2f * v;
        ebuf[(size_t)e * H + h] = v;
        atomicMax(&emax[(size_t)d * H + h], fenc(v));
    }
}

// Pass B: ebuf = exp(e - max[dst]); denom[dst] += ebuf
__global__ void edge_exp(const int* __restrict__ ei,
                         float* __restrict__ ebuf,
                         const unsigned* __restrict__ emax,
                         float* __restrict__ denom, int H) {
    int e = blockIdx.x * blockDim.x + threadIdx.x;
    if (e >= ETOT) return;
    int s, d; edge_sd(ei, e, s, d);
    for (int h = 0; h < H; ++h) {
        float ex = __expf(ebuf[(size_t)e * H + h] - fdec(emax[(size_t)d * H + h]));
        ebuf[(size_t)e * H + h] = ex;
        atomicAdd(&denom[(size_t)d * H + h], ex);
    }
}

// Pass C: out[dst] += alpha * hlin[src]  — one wave per edge, 32 lanes x HC/32
__global__ __launch_bounds__(256)
void edge_messages(const int* __restrict__ ei,
                   const float* __restrict__ ebuf,
                   const float* __restrict__ denom,
                   const float* __restrict__ hsrc,
                   float* __restrict__ out, int H) {
    int e    = blockIdx.x * 8 + (threadIdx.x >> 5);
    int lane = threadIdx.x & 31;
    if (e >= ETOT) return;
    int s, d; edge_sd(ei, e, s, d);
    float alpha[4];
    for (int h = 0; h < H; ++h)
        alpha[h] = ebuf[(size_t)e * H + h] /
                   (denom[(size_t)d * H + h] + 1e-16f);
    const int HC = H * 64;
    const float* hp = hsrc + (size_t)s * HC;
    float* op = out + (size_t)d * HC;
    for (int j = 0; j < HC; j += 32) {
        int c = j + lane;
        atomicAdd(&op[c], alpha[c >> 6] * hp[c]);
    }
}

// conv1 epilogue: h = elu(acc + b), emitted as f16 for GEMM2
__global__ void bias_elu_to_f16(const float* __restrict__ acc,
                                const float* __restrict__ b,
                                half_t* __restrict__ out, int n, int C) {
    int i = blockIdx.x * blockDim.x + threadIdx.x;
    if (i >= n) return;
    out[i] = (half_t)elu_f(acc[i] + b[i % C]);
}

// conv2 epilogue: in-place f32
__global__ void bias_elu_inplace(float* __restrict__ acc,
                                 const float* __restrict__ b, int n, int C) {
    int i = blockIdx.x * blockDim.x + threadIdx.x;
    if (i >= n) return;
    acc[i] = elu_f(acc[i] + b[i % C]);
}

// ------------------------- global attention pooling -------------------------

__global__ void gate_scores(const float* __restrict__ h,
                            const float* __restrict__ Wg,
                            const float* __restrict__ bg,
                            const int* __restrict__ batch,
                            float* __restrict__ att,
                            unsigned* __restrict__ gmax) {
    int n = blockIdx.x * blockDim.x + threadIdx.x;
    if (n >= NN) return;
    const float* hp = h + (size_t)n * HID;
    float a = bg[0];
#pragma unroll 8
    for (int c = 0; c < HID; ++c) a += hp[c] * Wg[c];
    att[n] = a;
    atomicMax(&gmax[batch[n]], fenc(a));
}

__global__ void gate_exp(float* __restrict__ att,
                         const unsigned* __restrict__ gmax,
                         float* __restrict__ gden,
                         const int* __restrict__ batch) {
    int n = blockIdx.x * blockDim.x + threadIdx.x;
    if (n >= NN) return;
    int g = batch[n];
    float ex = __expf(att[n] - fdec(gmax[g]));
    att[n] = ex;
    atomicAdd(&gden[g], ex);
}

__global__ __launch_bounds__(256)
void gate_pool(const float* __restrict__ h, const float* __restrict__ att,
               const float* __restrict__ gden, const int* __restrict__ batch,
               float* __restrict__ pooled) {
    int n    = blockIdx.x * 8 + (threadIdx.x >> 5);
    int lane = threadIdx.x & 31;
    if (n >= NN) return;
    int g = batch[n];
    float coeff = att[n] / (gden[g] + 1e-16f);
    const float* hp = h + (size_t)n * HID;
    float* pp = pooled + (size_t)g * HID;
    atomicAdd(&pp[lane],      coeff * hp[lane]);
    atomicAdd(&pp[lane + 32], coeff * hp[lane + 32]);
}

// ------------------------- classifier head ----------------------------------

__global__ __launch_bounds__(32)
void classifier(const float* __restrict__ pooled,
                const float* __restrict__ Wc1, const float* __restrict__ bc1,
                const float* __restrict__ Wc2, const float* __restrict__ bc2,
                float* __restrict__ out) {
    int g = blockIdx.x, lane = threadIdx.x;
    const float* pp = pooled + (size_t)g * HID;
    float acc = bc1[lane];
#pragma unroll 8
    for (int c = 0; c < HID; ++c) acc += pp[c] * Wc1[c * 32 + lane];
    acc = acc > 0.0f ? acc : 0.0f;
    __shared__ float hid[32];
    hid[lane] = acc;
    __syncthreads();
    if (lane < 2) {
        float o = bc2[lane];
#pragma unroll
        for (int j = 0; j < 32; ++j) o += hid[j] * Wc2[j * 2 + lane];
        out[g * 2 + lane] = o;
    }
}

// ------------------------- launch -------------------------------------------

static inline size_t align256(size_t x) { return (x + 255) & ~(size_t)255; }

extern "C" void kernel_launch(void* const* d_in, const int* in_sizes, int n_in,
                              void* d_out, int out_size, void* d_ws, size_t ws_size,
                              hipStream_t stream) {
    const float* x     = (const float*)d_in[0];
    const int*   ei    = (const int*)  d_in[1];
    const int*   batch = (const int*)  d_in[2];
    const float* W1    = (const float*)d_in[3];
    const float* as1   = (const float*)d_in[4];
    const float* ad1   = (const float*)d_in[5];
    const float* b1    = (const float*)d_in[6];
    const float* W2    = (const float*)d_in[7];
    const float* as2   = (const float*)d_in[8];
    const float* ad2   = (const float*)d_in[9];
    const float* b2    = (const float*)d_in[10];
    const float* Wg    = (const float*)d_in[11];
    const float* bg    = (const float*)d_in[12];
    const float* Wc1   = (const float*)d_in[13];
    const float* bc1   = (const float*)d_in[14];
    const float* Wc2   = (const float*)d_in[15];
    const float* bc2   = (const float*)d_in[16];
    float* out = (float*)d_out;

    // ---- workspace layout (regions reused across phases) ----
    char* ws = (char*)d_ws;
    size_t off = 0;
    auto take = [&](size_t bytes) { char* p = ws + off; off += align256(bytes); return p; };

    half_t*   xh     = (half_t*)  take((size_t)NN * FIN * 2);  // later: h1 f16
    half_t*   w1t    = (half_t*)  take((size_t)HC1 * FIN * 2);
    half_t*   w2t    = (half_t*)  take((size_t)HID * HC1 * 2);
    float*    hlin   = (float*)   take((size_t)NN * HC1 * 4);  // gemm1 out; later gemm2 out
    float*    outacc = (float*)   take((size_t)NN * HC1 * 4);  // conv1 acc; later conv2 acc
    float*    ebuf   = (float*)   take((size_t)ETOT * HEADS * 4);
    float*    asrc1  = (float*)   take((size_t)NN * HEADS * 4);
    float*    adst1  = (float*)   take((size_t)NN * HEADS * 4);
    unsigned* emax1  = (unsigned*)take((size_t)NN * HEADS * 4);
    float*    den1   = (float*)   take((size_t)NN * HEADS * 4);
    float*    asrc2  = (float*)   take((size_t)NN * 4);
    float*    adst2  = (float*)   take((size_t)NN * 4);
    unsigned* emax2  = (unsigned*)take((size_t)NN * 4);
    float*    den2   = (float*)   take((size_t)NN * 4);
    float*    att    = (float*)   take((size_t)NN * 4);
    unsigned* gmax   = (unsigned*)take((size_t)NG * 4);
    float*    gden   = (float*)   take((size_t)NG * 4);
    float*    pooled = (float*)   take((size_t)NG * HID * 4);
    (void)ws_size; (void)n_in; (void)in_sizes; (void)out_size;

    const int T = 256;
    const int EBLK = (ETOT + T - 1) / T;       // edge thread-per-edge blocks
    const int EWAV = (ETOT + 7) / 8;           // edge wave-per-edge blocks

    // ---- conv1 ----
    cvt_f32_to_f16<<<(NN * FIN + T - 1) / T, T, 0, stream>>>(x, xh, NN * FIN);
    transpose_to_f16<<<(FIN * HC1 + T - 1) / T, T, 0, stream>>>(W1, w1t, FIN, HC1);
    // 16x64 strip per wave: grid.y = 256 / (16*4) = 4
    wmma_gemm_f16<FIN, 4><<<dim3(NN / 16, HC1 / 64), 32, 0, stream>>>(xh, w1t, hlin, HC1);

    node_att_dots<<<(NN * HEADS + T - 1) / T, T, 0, stream>>>(hlin, as1, ad1, asrc1, adst1, HEADS);

    hipMemsetAsync(emax1, 0, (size_t)NN * HEADS * 4, stream);
    hipMemsetAsync(den1,  0, (size_t)NN * HEADS * 4, stream);
    hipMemsetAsync(outacc, 0, (size_t)NN * HC1 * 4, stream);

    edge_logits<<<EBLK, T, 0, stream>>>(ei, asrc1, adst1, ebuf, emax1, HEADS);
    edge_exp   <<<EBLK, T, 0, stream>>>(ei, ebuf, emax1, den1, HEADS);
    edge_messages<<<EWAV, T, 0, stream>>>(ei, ebuf, den1, hlin, outacc, HEADS);

    // epilogue: h1 = elu(acc + b1) -> f16 (reuse xh region)
    bias_elu_to_f16<<<(NN * HC1 + T - 1) / T, T, 0, stream>>>(outacc, b1, xh, NN * HC1, HC1);

    // ---- conv2 ----
    transpose_to_f16<<<(HC1 * HID + T - 1) / T, T, 0, stream>>>(W2, w2t, HC1, HID);
    // N = 64: one 16x64 strip per M tile: grid.y = 1
    wmma_gemm_f16<HC1, 4><<<dim3(NN / 16, HID / 64), 32, 0, stream>>>(xh, w2t, hlin, HID);

    node_att_dots<<<(NN + T - 1) / T, T, 0, stream>>>(hlin, as2, ad2, asrc2, adst2, 1);

    hipMemsetAsync(emax2, 0, (size_t)NN * 4, stream);
    hipMemsetAsync(den2,  0, (size_t)NN * 4, stream);
    hipMemsetAsync(outacc, 0, (size_t)NN * HID * 4, stream);

    edge_logits<<<EBLK, T, 0, stream>>>(ei, asrc2, adst2, ebuf, emax2, 1);
    edge_exp   <<<EBLK, T, 0, stream>>>(ei, ebuf, emax2, den2, 1);
    edge_messages<<<EWAV, T, 0, stream>>>(ei, ebuf, den2, hlin, outacc, 1);

    bias_elu_inplace<<<(NN * HID + T - 1) / T, T, 0, stream>>>(outacc, b2, NN * HID, HID);

    // ---- global attention pooling ----
    hipMemsetAsync(gmax, 0, (size_t)NG * 4, stream);
    hipMemsetAsync(gden, 0, (size_t)NG * 4, stream);
    hipMemsetAsync(pooled, 0, (size_t)NG * HID * 4, stream);

    gate_scores<<<(NN + T - 1) / T, T, 0, stream>>>(outacc, Wg, bg, batch, att, gmax);
    gate_exp   <<<(NN + T - 1) / T, T, 0, stream>>>(att, gmax, gden, batch);
    gate_pool  <<<(NN + 7) / 8, T, 0, stream>>>(outacc, att, gden, batch, pooled);

    // ---- classifier ----
    classifier<<<NG, 32, 0, stream>>>(pooled, Wc1, bc1, Wc2, bc2, out);
}
